// SE3EquivariantEncoder_71803263254751
// MI455X (gfx1250) — matrix operand order repaired and checked by reading, not compiled, not run on hardware
//
#include <hip/hip_runtime.h>
#include <hip/hip_bf16.h>
#include <math.h>

#define NN    16384
#define EE    524288
#define BB    8
#define F_IN  26
#define DD    128
#define HH    256
#define OUTD  64
#define LL    4
#define NRBF  16

typedef __attribute__((ext_vector_type(16))) _Float16 v16h;
typedef __attribute__((ext_vector_type(8)))  _Float16 v8h;
typedef __attribute__((ext_vector_type(8)))  float    v8f;

__device__ __forceinline__ float silu_f(float x) { return x / (1.0f + __expf(-x)); }

__device__ __forceinline__ v8f wmma_f16(v16h a, v16h b, v8f c) {
    return __builtin_amdgcn_wmma_f32_16x16x32_f16(false, a, false, b, (short)0, c, false, false);
}

// A fragment: 16x32 f16 tile, rowp = pointer to this lane's row (M = lane&15).
// element e -> K = kbase + half*16 + e
__device__ __forceinline__ v16h frag_a(const _Float16* rowp, int kbase, int half) {
    v8h lo = *(const v8h*)(rowp + kbase + half * 16);
    v8h hi = *(const v8h*)(rowp + kbase + half * 16 + 8);
    v16h a;
#pragma unroll
    for (int i = 0; i < 8; ++i) { a[i] = lo[i]; a[i + 8] = hi[i]; }
    return a;
}

// B fragment: 32x16 f16 tile. lane = col N (=lane&15), elements e -> K = kbase + half*16 + e
__device__ __forceinline__ v16h frag_b(const _Float16* p) {
    v8h lo = *(const v8h*)(p);
    v8h hi = *(const v8h*)(p + 8);
    v16h b;
#pragma unroll
    for (int i = 0; i < 8; ++i) { b[i] = lo[i]; b[i + 8] = hi[i]; }
    return b;
}

// ---------------- weight conversion (transpose to N-major, f16, pad K) ----------------
#define S1 (LL*256*288)  // msg_W1t: (L, 256n, 288k)  src (L,272,256)
#define S2 (LL*128*256)  // msg_W2t: (L, 128n, 256k)  src (L,256,128)
#define S3 (LL*256*256)  // upd_W1t: (L, 256n, 256k)  src (L,256,256)
#define S4 (LL*128*256)  // upd_W2t: (L, 128n, 256k)  src (L,256,128)

__global__ __launch_bounds__(256) void convert_weights_kernel(
    const float* __restrict__ mW1, const float* __restrict__ mW2,
    const float* __restrict__ uW1, const float* __restrict__ uW2,
    _Float16* __restrict__ mW1t, _Float16* __restrict__ mW2t,
    _Float16* __restrict__ uW1t, _Float16* __restrict__ uW2t)
{
    long i = (long)blockIdx.x * 256 + threadIdx.x;
    if (i < S1) {
        int l = (int)(i / (256 * 288)); int rem = (int)(i % (256 * 288));
        int n = rem / 288, k = rem % 288;
        mW1t[i] = (k < 272) ? (_Float16)mW1[((size_t)l * 272 + k) * 256 + n] : (_Float16)0.0f;
        return;
    }
    i -= S1;
    if (i < S2) {
        int l = (int)(i / (128 * 256)); int rem = (int)(i % (128 * 256));
        int n = rem / 256, k = rem % 256;
        mW2t[i] = (_Float16)mW2[((size_t)l * 256 + k) * 128 + n];
        return;
    }
    i -= S2;
    if (i < S3) {
        int l = (int)(i / (256 * 256)); int rem = (int)(i % (256 * 256));
        int n = rem / 256, k = rem % 256;
        uW1t[i] = (_Float16)uW1[((size_t)l * 256 + k) * 256 + n];
        return;
    }
    i -= S3;
    if (i < S4) {
        int l = (int)(i / (128 * 256)); int rem = (int)(i % (128 * 256));
        int n = rem / 256, k = rem % 256;
        uW2t[i] = (_Float16)uW2[((size_t)l * 256 + k) * 128 + n];
    }
}

// ---------------- edge sort by dst (counting sort; dst is layer-invariant) ----------------
__global__ __launch_bounds__(256) void hist_kernel(const int* __restrict__ dst, int* __restrict__ hist) {
    int e = blockIdx.x * 256 + threadIdx.x;
    if (e < EE) atomicAdd(&hist[dst[e]], 1);
}

__global__ __launch_bounds__(1024) void scan_kernel(const int* __restrict__ hist, int* __restrict__ cursor) {
    __shared__ int part[1024];
    int t = threadIdx.x;
    int loc[16];
    int s = 0;
#pragma unroll
    for (int i = 0; i < 16; ++i) { loc[i] = s; s += hist[t * 16 + i]; }
    part[t] = s;
    __syncthreads();
    for (int off = 1; off < 1024; off <<= 1) {
        int v = (t >= off) ? part[t - off] : 0;
        __syncthreads();
        part[t] += v;
        __syncthreads();
    }
    int base = part[t] - s;  // exclusive prefix
#pragma unroll
    for (int i = 0; i < 16; ++i) cursor[t * 16 + i] = base + loc[i];
}

__global__ __launch_bounds__(256) void scatter_kernel(const int* __restrict__ dst,
                                                      int* __restrict__ cursor, int* __restrict__ eperm) {
    int e = blockIdx.x * 256 + threadIdx.x;
    if (e < EE) {
        int p = atomicAdd(&cursor[dst[e]], 1);
        eperm[p] = e;
    }
}

// ---------------- node embedding MLP (tiny; plain VALU) ----------------
__global__ __launch_bounds__(128) void embed_kernel(
    const float* __restrict__ nf, const float* __restrict__ W1, const float* __restrict__ b1,
    const float* __restrict__ W2, const float* __restrict__ b2,
    float* __restrict__ h32, _Float16* __restrict__ h16)
{
    int wid = threadIdx.x >> 5, lane = threadIdx.x & 31;
    int node = blockIdx.x * 4 + wid;
    __shared__ float hid[4][DD];
    const float* x = nf + (size_t)node * F_IN;
#pragma unroll
    for (int jj = 0; jj < 4; ++jj) {
        int j = lane + jj * 32;
        float s = b1[j];
        for (int i = 0; i < F_IN; ++i) s += x[i] * W1[i * DD + j];
        hid[wid][j] = silu_f(s);
    }
    __syncthreads();
#pragma unroll
    for (int jj = 0; jj < 4; ++jj) {
        int j = lane + jj * 32;
        float s = b2[j];
        for (int i = 0; i < DD; ++i) s += hid[wid][i] * W2[i * DD + j];
        h32[(size_t)node * DD + j] = s;
        h16[(size_t)node * DD + j] = (_Float16)s;
    }
}

// ---------------- edge RBF features -> f16 ----------------
__global__ __launch_bounds__(256) void rbf_kernel(
    const float* __restrict__ pos, const int* __restrict__ src, const int* __restrict__ dst,
    _Float16* __restrict__ ef)
{
    int e = blockIdx.x * 256 + threadIdx.x;
    if (e >= EE) return;
    int s = src[e], d = dst[e];
    float dx = pos[(size_t)d * 3 + 0] - pos[(size_t)s * 3 + 0];
    float dy = pos[(size_t)d * 3 + 1] - pos[(size_t)s * 3 + 1];
    float dz = pos[(size_t)d * 3 + 2] - pos[(size_t)s * 3 + 2];
    float dist = sqrtf(dx * dx + dy * dy + dz * dz + 1e-12f);
    float env = (dist < 10.0f) ? 0.5f * (__cosf(3.14159265358979323846f * dist * 0.1f) + 1.0f) : 0.0f;
    const float w = 10.0f / (float)NRBF;           // width = 0.625
    const float inv2w2 = 1.0f / (2.0f * w * w);
    _Float16* out = ef + (size_t)e * NRBF;
#pragma unroll
    for (int i = 0; i < NRBF; ++i) {
        float c = (10.0f / 15.0f) * (float)i;      // linspace(0, 10, 16)
        float t = dist - c;
        out[i] = (_Float16)(__expf(-t * t * inv2w2) * env);
    }
}

// ---------------- message MLP + run-combined scatter-add (dominant kernel, WMMA) ----------------
// one wave handles MT*16 = 32 edges (sorted by dst); B fragments reused across both M-tiles
#define MT 2
__global__ __launch_bounds__(32) void msg_kernel(
    const _Float16* __restrict__ h16, const _Float16* __restrict__ ef16,
    const int* __restrict__ src, const int* __restrict__ dst,
    const int* __restrict__ eperm,
    const _Float16* __restrict__ W1t, const float* __restrict__ b1,
    const _Float16* __restrict__ W2t, const float* __restrict__ b2,
    float* __restrict__ agg)
{
    __shared__ __align__(32) _Float16 xbuf[MT * 16][288];   // [edge row][272 feat + pad]
    __shared__ __align__(32) _Float16 hidbuf[MT * 16][256];
    __shared__ int sidx[MT * 16];
    __shared__ int didx[MT * 16];
    __shared__ int epr[MT * 16];

    int lane = threadIdx.x & 31;
    int e0 = blockIdx.x * (MT * 16);

    int ep = eperm[e0 + lane];
    epr[lane] = ep;
    sidx[lane] = src[ep];
    didx[lane] = dst[ep];
    __syncthreads();

    // stage X = [h[src](128) | h[dst](128) | rbf(16) | zeros(16)]
    for (int c = lane; c < MT * 16 * 36; c += 32) {
        int row = c / 36, cc = c % 36;
        int f0 = cc * 8;
        uint4 v = make_uint4(0u, 0u, 0u, 0u);
        if (f0 < 128)       v = *(const uint4*)(h16 + (size_t)sidx[row] * DD + f0);
        else if (f0 < 256)  v = *(const uint4*)(h16 + (size_t)didx[row] * DD + (f0 - 128));
        else if (f0 < 272)  v = *(const uint4*)(ef16 + (size_t)epr[row] * NRBF + (f0 - 256));
        *(uint4*)(&xbuf[row][f0]) = v;
    }
    __syncthreads();

    int half = lane >> 4, nl = lane & 15;

    // hoist A fragments for both M-tiles (held in VGPRs across the whole N loop)
    v16h a0[9], a1[9];
#pragma unroll
    for (int kc = 0; kc < 9; ++kc) {
        a0[kc] = frag_a(&xbuf[nl][0], kc * 32, half);
        a1[kc] = frag_a(&xbuf[16 + nl][0], kc * 32, half);
    }

    // GEMM1: (32x288) @ (288x256) -> silu -> hidbuf
#pragma unroll 1
    for (int nt = 0; nt < 16; ++nt) {
        v8f c0 = {}, c1 = {};
#pragma unroll
        for (int kc = 0; kc < 9; ++kc) {
            v16h b = frag_b(W1t + (size_t)(nt * 16 + nl) * 288 + kc * 32 + half * 16);
            c0 = wmma_f16(a0[kc], b, c0);
            c1 = wmma_f16(a1[kc], b, c1);
        }
        float bias = b1[nt * 16 + nl];
#pragma unroll
        for (int r = 0; r < 8; ++r) {
            hidbuf[r + half * 8][nt * 16 + nl]      = (_Float16)silu_f(c0[r] + bias);
            hidbuf[16 + r + half * 8][nt * 16 + nl] = (_Float16)silu_f(c1[r] + bias);
        }
    }
    __syncthreads();

    v16h g0[8], g1[8];
#pragma unroll
    for (int kc = 0; kc < 8; ++kc) {
        g0[kc] = frag_a(&hidbuf[nl][0], kc * 32, half);
        g1[kc] = frag_a(&hidbuf[16 + nl][0], kc * 32, half);
    }

    // GEMM2: (32x256) @ (256x128) -> +bias -> run-combined atomic scatter
#pragma unroll 1
    for (int nt = 0; nt < 8; ++nt) {
        v8f c0 = {}, c1 = {};
#pragma unroll
        for (int kc = 0; kc < 8; ++kc) {
            v16h b = frag_b(W2t + (size_t)(nt * 16 + nl) * 256 + kc * 32 + half * 16);
            c0 = wmma_f16(g0[kc], b, c0);
            c1 = wmma_f16(g1[kc], b, c1);
        }
        float bias = b2[nt * 16 + nl];
        int col = nt * 16 + nl;
        {
            int base = half * 8;
            float vcur = c0[0] + bias;
            int dcur = didx[base];
#pragma unroll
            for (int r = 1; r < 8; ++r) {
                int dn = didx[base + r];
                float v = c0[r] + bias;
                if (dn == dcur) { vcur += v; }
                else { unsafeAtomicAdd(&agg[(size_t)dcur * DD + col], vcur); vcur = v; dcur = dn; }
            }
            unsafeAtomicAdd(&agg[(size_t)dcur * DD + col], vcur);
        }
        {
            int base = 16 + half * 8;
            float vcur = c1[0] + bias;
            int dcur = didx[base];
#pragma unroll
            for (int r = 1; r < 8; ++r) {
                int dn = didx[base + r];
                float v = c1[r] + bias;
                if (dn == dcur) { vcur += v; }
                else { unsafeAtomicAdd(&agg[(size_t)dcur * DD + col], vcur); vcur = v; dcur = dn; }
            }
            unsafeAtomicAdd(&agg[(size_t)dcur * DD + col], vcur);
        }
    }
}

// ---------------- node update MLP (WMMA) with residual ----------------
#define MW 2
__global__ __launch_bounds__(64) void upd_kernel(
    float* __restrict__ h32, _Float16* __restrict__ h16,
    const float* __restrict__ agg,
    const _Float16* __restrict__ W1t, const float* __restrict__ b1,
    const _Float16* __restrict__ W2t, const float* __restrict__ b2)
{
    __shared__ __align__(32) _Float16 ubuf[MW][16][256];    // [node row][h(128) | agg(128)]
    __shared__ __align__(32) _Float16 hidbuf[MW][16][256];

    int wid = threadIdx.x >> 5, lane = threadIdx.x & 31;
    int tile = blockIdx.x * MW + wid;
    int n0 = tile * 16;

    for (int c = lane; c < 16 * 32; c += 32) {
        int row = c / 32, cc = c % 32;
        int f0 = cc * 8;
        int node = n0 + row;
        if (f0 < 128) {
            *(uint4*)(&ubuf[wid][row][f0]) = *(const uint4*)(h16 + (size_t)node * DD + f0);
        } else {
            const float* ap = agg + (size_t)node * DD + (f0 - 128);
            float4 q0 = *(const float4*)(ap);
            float4 q1 = *(const float4*)(ap + 4);
            v8h v;
            v[0] = (_Float16)q0.x; v[1] = (_Float16)q0.y; v[2] = (_Float16)q0.z; v[3] = (_Float16)q0.w;
            v[4] = (_Float16)q1.x; v[5] = (_Float16)q1.y; v[6] = (_Float16)q1.z; v[7] = (_Float16)q1.w;
            *(v8h*)(&ubuf[wid][row][f0]) = v;
        }
    }
    __syncthreads();

    int half = lane >> 4, nl = lane & 15;
    const _Float16* arow = &ubuf[wid][nl][0];

#pragma unroll 1
    for (int nt = 0; nt < 16; ++nt) {
        v8f acc = {};
#pragma unroll
        for (int kc = 0; kc < 8; ++kc) {
            v16h a = frag_a(arow, kc * 32, half);
            v16h b = frag_b(W1t + (size_t)(nt * 16 + nl) * 256 + kc * 32 + half * 16);
            acc = wmma_f16(a, b, acc);
        }
        float bias = b1[nt * 16 + nl];
#pragma unroll
        for (int r = 0; r < 8; ++r) {
            hidbuf[wid][r + half * 8][nt * 16 + nl] = (_Float16)silu_f(acc[r] + bias);
        }
    }
    __syncthreads();

    const _Float16* arow2 = &hidbuf[wid][nl][0];
#pragma unroll 1
    for (int nt = 0; nt < 8; ++nt) {
        v8f acc = {};
#pragma unroll
        for (int kc = 0; kc < 8; ++kc) {
            v16h a = frag_a(arow2, kc * 32, half);
            v16h b = frag_b(W2t + (size_t)(nt * 16 + nl) * 256 + kc * 32 + half * 16);
            acc = wmma_f16(a, b, acc);
        }
        float bias = b2[nt * 16 + nl];
#pragma unroll
        for (int r = 0; r < 8; ++r) {
            int node = n0 + r + half * 8;
            int col = nt * 16 + nl;
            float res = h32[(size_t)node * DD + col] + acc[r] + bias;
            h32[(size_t)node * DD + col] = res;
            h16[(size_t)node * DD + col] = (_Float16)res;
        }
    }
}

// ---------------- readout ----------------
__global__ __launch_bounds__(256) void gacc_kernel(
    const float* __restrict__ h32, const int* __restrict__ batch,
    float* __restrict__ gsum, float* __restrict__ gcount)
{
    int t = blockIdx.x * 256 + threadIdx.x;  // NN*4 threads
    int node = t >> 2, ch = t & 3;
    if (node >= NN) return;
    int b = batch[node];
    const float* hr = h32 + (size_t)node * DD + ch * 32;
    float* gs = gsum + (size_t)b * DD + ch * 32;
    for (int i = 0; i < 32; ++i) unsafeAtomicAdd(&gs[i], hr[i]);
    if (ch == 0) unsafeAtomicAdd(&gcount[b], 1.0f);
}

__global__ __launch_bounds__(256) void readout_kernel(
    const float* __restrict__ gsum, const float* __restrict__ gcount,
    const float* __restrict__ W1, const float* __restrict__ b1,
    const float* __restrict__ W2, const float* __restrict__ b2,
    float* __restrict__ out)
{
    __shared__ float m[DD];
    __shared__ float hid[HH];
    int t = threadIdx.x;
    for (int g = 0; g < BB; ++g) {
        if (t < DD) m[t] = gsum[(size_t)g * DD + t] / fmaxf(gcount[g], 1.0f);
        __syncthreads();
        float s = b1[t];
        for (int i = 0; i < DD; ++i) s += m[i] * W1[i * HH + t];
        hid[t] = silu_f(s);
        __syncthreads();
        if (t < OUTD) {
            float o = b2[t];
            for (int i = 0; i < HH; ++i) o += hid[i] * W2[i * OUTD + t];
            out[(size_t)g * OUTD + t] = o;
        }
        __syncthreads();
    }
}

extern "C" void kernel_launch(void* const* d_in, const int* in_sizes, int n_in,
                              void* d_out, int out_size, void* d_ws, size_t ws_size,
                              hipStream_t stream) {
    const float* positions     = (const float*)d_in[0];
    const float* node_features = (const float*)d_in[1];
    const int*   edge_index    = (const int*)d_in[2];
    const int*   batch         = (const int*)d_in[3];
    const float* emb_W1 = (const float*)d_in[4];
    const float* emb_b1 = (const float*)d_in[5];
    const float* emb_W2 = (const float*)d_in[6];
    const float* emb_b2 = (const float*)d_in[7];
    const float* msg_W1 = (const float*)d_in[8];
    const float* msg_b1 = (const float*)d_in[9];
    const float* msg_W2 = (const float*)d_in[10];
    const float* msg_b2 = (const float*)d_in[11];
    const float* upd_W1 = (const float*)d_in[12];
    const float* upd_b1 = (const float*)d_in[13];
    const float* upd_W2 = (const float*)d_in[14];
    const float* upd_b2 = (const float*)d_in[15];
    const float* r_W1   = (const float*)d_in[16];
    const float* r_b1   = (const float*)d_in[17];
    const float* r_W2   = (const float*)d_in[18];
    const float* r_b2   = (const float*)d_in[19];

    const int* src = edge_index;
    const int* dst = edge_index + EE;

    char* ws = (char*)d_ws;
    size_t off = 0;
    auto alloc = [&](size_t bytes) -> void* {
        void* p = ws + off;
        off = (off + bytes + 255) & ~(size_t)255;
        return p;
    };
    float*    h32    = (float*)   alloc((size_t)NN * DD * 4);
    _Float16* h16    = (_Float16*)alloc((size_t)NN * DD * 2);
    _Float16* ef16   = (_Float16*)alloc((size_t)EE * NRBF * 2);
    float*    agg    = (float*)   alloc((size_t)NN * DD * 4);
    _Float16* mW1t   = (_Float16*)alloc((size_t)S1 * 2);
    _Float16* mW2t   = (_Float16*)alloc((size_t)S2 * 2);
    _Float16* uW1t   = (_Float16*)alloc((size_t)S3 * 2);
    _Float16* uW2t   = (_Float16*)alloc((size_t)S4 * 2);
    int*      hist   = (int*)     alloc((size_t)NN * 4);
    int*      cursor = (int*)     alloc((size_t)NN * 4);
    int*      eperm  = (int*)     alloc((size_t)EE * 4);
    float*    gsum   = (float*)   alloc((size_t)(BB * DD + BB) * 4);
    float*    gcount = gsum + BB * DD;

    // edge sort by dst (layer-invariant)
    hipMemsetAsync(hist, 0, (size_t)NN * 4, stream);
    hist_kernel<<<EE / 256, 256, 0, stream>>>(dst, hist);
    scan_kernel<<<1, 1024, 0, stream>>>(hist, cursor);
    scatter_kernel<<<EE / 256, 256, 0, stream>>>(dst, cursor, eperm);

    const long totW = (long)S1 + S2 + S3 + S4;
    convert_weights_kernel<<<(unsigned)((totW + 255) / 256), 256, 0, stream>>>(
        msg_W1, msg_W2, upd_W1, upd_W2, mW1t, mW2t, uW1t, uW2t);

    embed_kernel<<<NN / 4, 128, 0, stream>>>(node_features, emb_W1, emb_b1, emb_W2, emb_b2, h32, h16);
    rbf_kernel<<<EE / 256, 256, 0, stream>>>(positions, src, dst, ef16);

    for (int l = 0; l < LL; ++l) {
        hipMemsetAsync(agg, 0, (size_t)NN * DD * 4, stream);
        msg_kernel<<<EE / (MT * 16), 32, 0, stream>>>(
            h16, ef16, src, dst, eperm,
            mW1t + (size_t)l * 256 * 288, msg_b1 + (size_t)l * 256,
            mW2t + (size_t)l * 128 * 256, msg_b2 + (size_t)l * 128, agg);
        upd_kernel<<<(NN / 16) / MW, 32 * MW, 0, stream>>>(
            h32, h16, agg,
            uW1t + (size_t)l * 256 * 256, upd_b1 + (size_t)l * 256,
            uW2t + (size_t)l * 128 * 256, upd_b2 + (size_t)l * 128);
    }

    hipMemsetAsync(gsum, 0, (size_t)(BB * DD + BB) * 4, stream);
    gacc_kernel<<<NN * 4 / 256, 256, 0, stream>>>(h32, batch, gsum, gcount);
    readout_kernel<<<1, 256, 0, stream>>>(gsum, gcount, r_W1, r_b1, r_W2, r_b2, (float*)d_out);
}
